// BitLinear_10350871183908
// MI455X (gfx1250) — matrix operand order repaired and checked by reading, not compiled, not run on hardware
//
#include <hip/hip_runtime.h>
#include <stdint.h>

// ---------------------------------------------------------------------------
// BitLinear: out[m,n] = sum_k input[m,k] * sign(weight[n,k])
//   input : fp32 [8192, 4096]  (4 x 2048 flattened),  weight: fp32 [4096, 4096]
// Fast path (needs 167.8 MB workspace):
//   pass 1: convert input -> bf16 hi/lo pair, weight -> bf16 sign (exact)
//   pass 2: GEMM with v_wmma_f32_16x16x32_bf16; 64x64 wave tiles, tiles
//           filled with global_load_async_to_lds_b128 (ASYNCcnt), double LDS.
// Fallback: single kernel converting inline.
// ---------------------------------------------------------------------------

typedef __attribute__((ext_vector_type(16))) __bf16 v16bf;
typedef __attribute__((ext_vector_type(8)))  float  v8f;

#define M_DIM 8192
#define N_DIM 4096
#define K_DIM 4096
#define BM 128
#define BN 128
#define BK 32
#define LSTR 40                 // bf16 per LDS row: 32 data + 8 pad (80 B)
#define LDS_BUF_BYTES (BM * LSTR * 2)

struct alignas(16) U4 { uint32_t x, y, z, w; };
struct alignas(8)  U2 { uint32_t x, y; };
struct alignas(16) F4 { float x, y, z, w; };

union Frag { uint32_t u[8]; v16bf v; };

__device__ __forceinline__ uint32_t f2bf(float f) {
  uint32_t u = __builtin_bit_cast(uint32_t, f);
  uint32_t r = u + 0x7FFFu + ((u >> 16) & 1u);   // RNE
  return r >> 16;
}
__device__ __forceinline__ float bf2f(uint32_t h) {
  return __builtin_bit_cast(float, h << 16);
}
__device__ __forceinline__ void ldfrag(Frag& f, const uint16_t* p0, const uint16_t* p1) {
  const U4 q0 = *(const U4*)p0;
  const U4 q1 = *(const U4*)p1;
  f.u[0] = q0.x; f.u[1] = q0.y; f.u[2] = q0.z; f.u[3] = q0.w;
  f.u[4] = q1.x; f.u[5] = q1.y; f.u[6] = q1.z; f.u[7] = q1.w;
}

// ---------------- pass 1: conversions (memory-bound) -------------------------
__global__ __launch_bounds__(256)
void prep_input_kernel(const float* __restrict__ in,
                       uint16_t* __restrict__ hi, uint16_t* __restrict__ lo) {
  const size_t i4 = ((size_t)blockIdx.x * 256 + threadIdx.x) * 4;
  const F4 v = *(const F4*)(in + i4);
  const float a[4] = {v.x, v.y, v.z, v.w};
  uint32_t hb[4], lb[4];
#pragma unroll
  for (int j = 0; j < 4; ++j) {
    hb[j] = f2bf(a[j]);
    lb[j] = f2bf(a[j] - bf2f(hb[j]));
  }
  *(U2*)(hi + i4) = U2{hb[0] | (hb[1] << 16), hb[2] | (hb[3] << 16)};
  *(U2*)(lo + i4) = U2{lb[0] | (lb[1] << 16), lb[2] | (lb[3] << 16)};
}

__global__ __launch_bounds__(256)
void prep_weight_kernel(const float* __restrict__ w, uint16_t* __restrict__ sgn) {
  const size_t i4 = ((size_t)blockIdx.x * 256 + threadIdx.x) * 4;
  const F4 v = *(const F4*)(w + i4);
  const float a[4] = {v.x, v.y, v.z, v.w};
  uint32_t sb[4];
#pragma unroll
  for (int j = 0; j < 4; ++j) {
    const float s = (a[j] > 0.0f) ? 1.0f : ((a[j] < 0.0f) ? -1.0f : 0.0f);
    sb[j] = f2bf(s);            // exact: 0x3F80 / 0xBF80 / 0x0000
  }
  *(U2*)(sgn + i4) = U2{sb[0] | (sb[1] << 16), sb[2] | (sb[3] << 16)};
}

// ---------------- pass 2: WMMA GEMM, 64x64 wave tiles, async LDS fill --------
__global__ __launch_bounds__(128)
void bitlinear_wmma_async_kernel(const uint16_t* __restrict__ Ah,
                                 const uint16_t* __restrict__ Al,
                                 const uint16_t* __restrict__ Ws,
                                 float* __restrict__ out) {
  __shared__ alignas(16) uint16_t sAh[2][BM * LSTR];
  __shared__ alignas(16) uint16_t sAl[2][BM * LSTR];
  __shared__ alignas(16) uint16_t sB [2][BN * LSTR];

  const int tid  = threadIdx.x;          // 0..127, 4 waves
  const int lane = tid & 31;
  const int wave = tid >> 5;
  const int wm0  = (wave >> 1) * 64;     // 2x2 waves, 64x64 each
  const int wn0  = (wave &  1) * 64;
  const int bm0 = blockIdx.y * BM;
  const int bn0 = blockIdx.x * BN;

  // flat->LDS offset (low 32 bits of flat shared address == AS3 byte offset)
  const uint32_t ldsAh = (uint32_t)(uintptr_t)(void*)&sAh[0][0];
  const uint32_t ldsAl = (uint32_t)(uintptr_t)(void*)&sAl[0][0];
  const uint32_t ldsB  = (uint32_t)(uintptr_t)(void*)&sB [0][0];

  v8f acc[4][4] = {};
  const int kTiles = K_DIM / BK;

  // async tile fill: 512 16-byte chunks per array, 4 per thread per array
  auto asyncLoadTile = [&](int kt, int buf) {
    const int k0 = kt * BK;
    const uint32_t bufOff = (uint32_t)buf * LDS_BUF_BYTES;
#pragma unroll
    for (int c = 0; c < 4; ++c) {
      const int chunk = tid + c * 128;          // 0..511
      const int row = chunk >> 2;               // 0..127
      const int col = chunk & 3;                // 16B chunk in row
      const uint32_t ldsOff = bufOff + (uint32_t)row * (LSTR * 2) + (uint32_t)col * 16;
      const size_t gA = (size_t)(bm0 + row) * K_DIM + k0 + col * 8;
      const size_t gW = (size_t)(bn0 + row) * K_DIM + k0 + col * 8;
      const uint32_t dA = ldsAh + ldsOff;
      const uint32_t dL = ldsAl + ldsOff;
      const uint32_t dB = ldsB  + ldsOff;
      const uint64_t pA = (uint64_t)(uintptr_t)(Ah + gA);
      const uint64_t pL = (uint64_t)(uintptr_t)(Al + gA);
      const uint64_t pW = (uint64_t)(uintptr_t)(Ws + gW);
      asm volatile("global_load_async_to_lds_b128 %0, %1, off"
                   :: "v"(dA), "v"(pA) : "memory");
      asm volatile("global_load_async_to_lds_b128 %0, %1, off"
                   :: "v"(dL), "v"(pL) : "memory");
      asm volatile("global_load_async_to_lds_b128 %0, %1, off"
                   :: "v"(dB), "v"(pW) : "memory");
    }
  };

  asyncLoadTile(0, 0);
  asm volatile("s_wait_asynccnt 0" ::: "memory");
  __syncthreads();

  const int nlane = lane & 15;
  const int kbA = (lane >> 4) * 8;    // A frag: K kbA..+7 then kbA+16..+23
  const int kbB = (lane >> 4) * 8;    // B frag: K kbB..+7 then kbB+16..+23

  for (int kt = 0; kt < kTiles; ++kt) {
    const int cur = kt & 1;
    if (kt + 1 < kTiles) asyncLoadTile(kt + 1, cur ^ 1);

    // preload all fragments into distinct registers, then chain 32 WMMAs
    Frag bfv[4], ahv[4], alv[4];
#pragma unroll
    for (int tn = 0; tn < 4; ++tn) {
      const uint16_t* p = &sB[cur][(wn0 + tn * 16 + nlane) * LSTR + kbB];
      ldfrag(bfv[tn], p, p + 16);
    }
#pragma unroll
    for (int tm = 0; tm < 4; ++tm) {
      const int row = (wm0 + tm * 16 + nlane) * LSTR + kbA;
      { const uint16_t* p = &sAh[cur][row]; ldfrag(ahv[tm], p, p + 16); }
      { const uint16_t* p = &sAl[cur][row]; ldfrag(alv[tm], p, p + 16); }
    }
#pragma unroll
    for (int tm = 0; tm < 4; ++tm)
#pragma unroll
      for (int tn = 0; tn < 4; ++tn) {
        acc[tm][tn] = __builtin_amdgcn_wmma_f32_16x16x32_bf16(
            false, ahv[tm].v, false, bfv[tn].v, (short)0, acc[tm][tn], false, false);
        acc[tm][tn] = __builtin_amdgcn_wmma_f32_16x16x32_bf16(
            false, alv[tm].v, false, bfv[tn].v, (short)0, acc[tm][tn], false, false);
      }

    asm volatile("s_wait_asynccnt 0" ::: "memory");
    __syncthreads();
  }

  const int mhi = (lane >> 4) * 8;
#pragma unroll
  for (int tm = 0; tm < 4; ++tm)
#pragma unroll
    for (int tn = 0; tn < 4; ++tn) {
      const int gm = bm0 + wm0 + tm * 16 + mhi;
      const int gn = bn0 + wn0 + tn * 16 + nlane;
#pragma unroll
      for (int e = 0; e < 8; ++e)
        out[(size_t)(gm + e) * N_DIM + gn] = acc[tm][tn][e];
    }
}

// ---------------- fallback: inline-conversion kernel -------------------------
__global__ __launch_bounds__(256)
void bitlinear_wmma_kernel(const float* __restrict__ input,
                           const float* __restrict__ weight,
                           float* __restrict__ out) {
  __shared__ alignas(16) uint16_t sAh[2][BM * LSTR];
  __shared__ alignas(16) uint16_t sAl[2][BM * LSTR];
  __shared__ alignas(16) uint16_t sB [2][BN * LSTR];

  const int tid  = threadIdx.x;
  const int lane = tid & 31;
  const int wave = tid >> 5;
  const int wm0  = (wave >> 2) * 64;
  const int wn0  = (wave & 3) * 32;
  const int bm0 = blockIdx.y * BM;
  const int bn0 = blockIdx.x * BN;
  const int lr = tid >> 3;
  const int lc = (tid & 7) << 2;

  v8f acc[4][2] = {};
  const int kTiles = K_DIM / BK;

  auto loadTile = [&](int kt, int buf) {
    const int k0 = kt * BK;
#pragma unroll
    for (int p = 0; p < 4; ++p) {
      const int row = lr + p * 32;
      const F4 a = *(const F4*)(input  + (size_t)(bm0 + row) * K_DIM + k0 + lc);
      const F4 w = *(const F4*)(weight + (size_t)(bn0 + row) * K_DIM + k0 + lc);
      const float av[4] = {a.x, a.y, a.z, a.w};
      const float wv[4] = {w.x, w.y, w.z, w.w};
      uint32_t hb[4], lb[4], sb[4];
#pragma unroll
      for (int j = 0; j < 4; ++j) {
        hb[j] = f2bf(av[j]);
        lb[j] = f2bf(av[j] - bf2f(hb[j]));
        const float s = (wv[j] > 0.0f) ? 1.0f : ((wv[j] < 0.0f) ? -1.0f : 0.0f);
        sb[j] = f2bf(s);
      }
      const int off = row * LSTR + lc;
      *(U2*)&sAh[buf][off] = U2{hb[0] | (hb[1] << 16), hb[2] | (hb[3] << 16)};
      *(U2*)&sAl[buf][off] = U2{lb[0] | (lb[1] << 16), lb[2] | (lb[3] << 16)};
      *(U2*)&sB [buf][off] = U2{sb[0] | (sb[1] << 16), sb[2] | (sb[3] << 16)};
    }
    if (kt + 2 < kTiles) {
      __builtin_prefetch(input  + (size_t)(bm0 + lr) * K_DIM + kt * BK + 2 * BK + lc, 0, 0);
      __builtin_prefetch(weight + (size_t)(bn0 + lr) * K_DIM + kt * BK + 2 * BK + lc, 0, 0);
    }
  };

  loadTile(0, 0);
  __syncthreads();

  const int nlane = lane & 15;
  const int kbA = (lane >> 4) * 8;
  const int kbB = (lane >> 4) * 8;

  for (int kt = 0; kt < kTiles; ++kt) {
    const int cur = kt & 1;
    if (kt + 1 < kTiles) loadTile(kt + 1, cur ^ 1);

    Frag bf[2];
#pragma unroll
    for (int tn = 0; tn < 2; ++tn) {
      const uint16_t* p = &sB[cur][(wn0 + tn * 16 + nlane) * LSTR + kbB];
      ldfrag(bf[tn], p, p + 16);
    }
#pragma unroll
    for (int tm = 0; tm < 4; ++tm) {
      const int row = (wm0 + tm * 16 + nlane) * LSTR + kbA;
      Frag ah, al;
      { const uint16_t* p = &sAh[cur][row]; ldfrag(ah, p, p + 16); }
      { const uint16_t* p = &sAl[cur][row]; ldfrag(al, p, p + 16); }
#pragma unroll
      for (int tn = 0; tn < 2; ++tn) {
        acc[tm][tn] = __builtin_amdgcn_wmma_f32_16x16x32_bf16(
            false, ah.v, false, bf[tn].v, (short)0, acc[tm][tn], false, false);
        acc[tm][tn] = __builtin_amdgcn_wmma_f32_16x16x32_bf16(
            false, al.v, false, bf[tn].v, (short)0, acc[tm][tn], false, false);
      }
    }
    __syncthreads();
  }

  const int mhi = (lane >> 4) * 8;
#pragma unroll
  for (int tm = 0; tm < 4; ++tm)
#pragma unroll
    for (int tn = 0; tn < 2; ++tn) {
      const int gm = bm0 + wm0 + tm * 16 + mhi;
      const int gn = bn0 + wn0 + tn * 16 + nlane;
#pragma unroll
      for (int e = 0; e < 8; ++e)
        out[(size_t)(gm + e) * N_DIM + gn] = acc[tm][tn][e];
    }
}

// ---------------------------------------------------------------------------
extern "C" void kernel_launch(void* const* d_in, const int* in_sizes, int n_in,
                              void* d_out, int out_size, void* d_ws, size_t ws_size,
                              hipStream_t stream) {
  (void)in_sizes; (void)n_in; (void)out_size;
  const float* input  = (const float*)d_in[0];
  const float* weight = (const float*)d_in[1];
  float* out = (float*)d_out;

  const size_t bytesA = (size_t)M_DIM * K_DIM * 2;   // 67.1 MB each
  const size_t bytesW = (size_t)N_DIM * K_DIM * 2;   // 33.6 MB
  const size_t need   = 2 * bytesA + bytesW;         // 167.8 MB

  dim3 grid(N_DIM / BN, M_DIM / BM);                 // (32, 64)

  if (ws_size >= need) {
    uint16_t* Ah = (uint16_t*)d_ws;
    uint16_t* Al = (uint16_t*)((char*)d_ws + bytesA);
    uint16_t* Ws = (uint16_t*)((char*)d_ws + 2 * bytesA);
    prep_input_kernel <<<(M_DIM * (size_t)K_DIM) / (256 * 4), 256, 0, stream>>>(input, Ah, Al);
    prep_weight_kernel<<<(N_DIM * (size_t)K_DIM) / (256 * 4), 256, 0, stream>>>(weight, Ws);
    bitlinear_wmma_async_kernel<<<grid, 128, 0, stream>>>(Ah, Al, Ws, out);
  } else {
    bitlinear_wmma_kernel<<<grid, 256, 0, stream>>>(input, weight, out);
  }
}
